// Attention_33011118637459
// MI455X (gfx1250) — matrix operand order repaired and checked by reading, not compiled, not run on hardware
//
#include <hip/hip_runtime.h>
#include <hip/hip_bf16.h>

typedef __attribute__((ext_vector_type(16))) _Float16 v16h;
typedef __attribute__((ext_vector_type(8)))  _Float16 v8h;
typedef __attribute__((ext_vector_type(4)))  _Float16 v4h;
typedef __attribute__((ext_vector_type(8)))  float    v8f;
typedef __attribute__((ext_vector_type(4)))  float    v4f;

#define HD __device__ __forceinline__

static constexpr int B_  = 2;
static constexpr int NQ  = 2048;
static constexpr int NKV = 2048;
static constexpr int C_  = 768;
static constexpr int H_  = 12;
static constexpr int D_  = 64;
static constexpr float SCALE = 0.125f;   // D^-0.5 = 1/8 (exact in f16)

// ---------------------------------------------------------------------------
// WMMA fragment helpers (CDNA5 16x16x32 f16 layouts, wave32)
// A (16Mx32K): lane l(0..15)+16g holds row M=l; halves 0..7 -> K = 8g+0..7,
//              halves 8..15 -> K = 16+8g+0..7   => two contiguous b128 loads.
// B (32Kx16N): lane l+16g holds column N=l; halves -> K as in A  => 16 strided loads.
// C/D (16Mx16N f32): reg v, lanes 0..15: M=v, N=lane; lanes 16..31: M=v+8.
// ---------------------------------------------------------------------------
HD v16h load_a_frag(const _Float16* __restrict__ A, int lda, int row0, int k0, int lane) {
  int l = lane & 15, g = lane >> 4;
  const _Float16* p = A + (size_t)(row0 + l) * lda + k0 + g * 8;
  v8h x = *(const v8h*)p;          // K = k0 + 8g + 0..7
  v8h y = *(const v8h*)(p + 16);   // K = k0 + 16 + 8g + 0..7
  v16h a;
#pragma unroll
  for (int i = 0; i < 8; ++i) { a[i] = x[i]; a[i + 8] = y[i]; }
  return a;
}

HD v16h load_b_frag(const _Float16* __restrict__ Bm, int ldb, int k0, int n0, int lane) {
  int l = lane & 15, g = lane >> 4;
  const _Float16* p = Bm + (size_t)(k0 + g * 8) * ldb + n0 + l;
  v16h b;
#pragma unroll
  for (int i = 0; i < 8; ++i) { b[i] = p[(size_t)i * ldb]; b[i + 8] = p[(size_t)(i + 16) * ldb]; }
  return b;
}

HD v8f wmma_f16(v16h a, v16h b, v8f c) {
  return __builtin_amdgcn_wmma_f32_16x16x32_f16(false, a, false, b, (short)0, c,
                                                false, false);
}

// ---------------------------------------------------------------------------
// fp32 -> f16 conversion (4 elements / thread)
// ---------------------------------------------------------------------------
__global__ void cvt_kernel(const float* __restrict__ s, _Float16* __restrict__ d, int n) {
  int i = (blockIdx.x * blockDim.x + threadIdx.x) * 4;
  if (i < n) {
    v4f x = *(const v4f*)(s + i);
    v4h y;
#pragma unroll
    for (int j = 0; j < 4; ++j) y[j] = (_Float16)x[j];
    *(v4h*)(d + i) = y;
  }
}

// ---------------------------------------------------------------------------
// Shared GEMM core: each wave computes a 32x32 tile of A[MxK] * W[KxN]
// ---------------------------------------------------------------------------
HD void gemm_core(const _Float16* __restrict__ A, const _Float16* __restrict__ W,
                  int K, int N, int row0, int col0, int lane, v8f acc[2][2]) {
  for (int k = 0; k < K; k += 32) {
    v16h a0 = load_a_frag(A, K, row0,      k, lane);
    v16h a1 = load_a_frag(A, K, row0 + 16, k, lane);
    v16h b0 = load_b_frag(W, N, k, col0,      lane);
    v16h b1 = load_b_frag(W, N, k, col0 + 16, lane);
    acc[0][0] = wmma_f16(a0, b0, acc[0][0]);
    acc[0][1] = wmma_f16(a0, b1, acc[0][1]);
    acc[1][0] = wmma_f16(a1, b0, acc[1][0]);
    acc[1][1] = wmma_f16(a1, b1, acc[1][1]);
  }
}

// ---------------------------------------------------------------------------
// Q projection:  q16[4096x768] * Wq16[768x768] -> Qt [BH][64][NQ]  (transposed,
// pre-scaled by 1/8).  Transposed store = contiguous b128 per lane.
// ---------------------------------------------------------------------------
__global__ void gemm_q_kernel(const _Float16* __restrict__ A, const _Float16* __restrict__ W,
                              _Float16* __restrict__ Qt) {
  const int N = C_, K = C_;
  int wid  = blockIdx.x * (blockDim.x >> 5) + (threadIdx.x >> 5);
  int tnn  = N / 32;
  int row0 = (wid / tnn) * 32, col0 = (wid % tnn) * 32;
  int lane = threadIdx.x & 31;
  v8f acc[2][2] = {};
  gemm_core(A, W, K, N, row0, col0, lane, acc);

  int l = lane & 15, g = lane >> 4;
#pragma unroll
  for (int rt = 0; rt < 2; ++rt)
#pragma unroll
    for (int ct = 0; ct < 2; ++ct) {
      int c  = col0 + ct * 16 + l;
      int hh = c >> 6, d = c & 63;
      int r  = row0 + rt * 16 + 8 * g;     // 8 consecutive rows (q) per lane
      int b  = r >> 11, q = r & 2047;
      v8h pk;
#pragma unroll
      for (int i = 0; i < 8; ++i) pk[i] = (_Float16)(acc[rt][ct][i] * SCALE);
      *(v8h*)(Qt + ((size_t)((b * H_ + hh) * D_ + d) * NQ + q)) = pk;
    }
}

// ---------------------------------------------------------------------------
// KV projection: kv16[4096x768] * Wkv16[768x1536] -> K,V row-major [BH][NKV][64]
// ---------------------------------------------------------------------------
__global__ void gemm_kv_kernel(const _Float16* __restrict__ A, const _Float16* __restrict__ W,
                               _Float16* __restrict__ Kb, _Float16* __restrict__ Vb) {
  const int N = 2 * C_, K = C_;
  int wid  = blockIdx.x * (blockDim.x >> 5) + (threadIdx.x >> 5);
  int tnn  = N / 32;
  int row0 = (wid / tnn) * 32, col0 = (wid % tnn) * 32;
  int lane = threadIdx.x & 31;
  v8f acc[2][2] = {};
  gemm_core(A, W, K, N, row0, col0, lane, acc);

  int l = lane & 15, g = lane >> 4;
#pragma unroll
  for (int rt = 0; rt < 2; ++rt)
#pragma unroll
    for (int ct = 0; ct < 2; ++ct) {
      int c   = col0 + ct * 16 + l;        // 0..1535
      int s   = c >= C_;                   // 0 -> K, 1 -> V
      int rem = c - s * C_;
      int hh  = rem >> 6, d = rem & 63;
      _Float16* dst = s ? Vb : Kb;
#pragma unroll
      for (int v = 0; v < 8; ++v) {
        int r  = row0 + rt * 16 + 8 * g + v;
        int b  = r >> 11, kv = r & 2047;
        dst[((size_t)((b * H_ + hh) * NKV + kv)) * D_ + d] = (_Float16)acc[rt][ct][v];
      }
    }
}

// ---------------------------------------------------------------------------
// Flash attention. One wave per (b,h,16-query tile).
// Trick: compute St = K_tile * Q^T.  St's C-layout (lane=q, regs=kv split by
// half-wave) is exactly the A-fragment layout of P (16q x 32kv), so the
// softmax output feeds the P*V WMMA with no cross-lane movement.
// ---------------------------------------------------------------------------
__global__ void attn_kernel(const _Float16* __restrict__ Qt, const _Float16* __restrict__ Kb,
                            const _Float16* __restrict__ Vb, _Float16* __restrict__ X) {
  int wid = blockIdx.x * (blockDim.x >> 5) + (threadIdx.x >> 5);
  int bh  = wid / (NQ / 16);
  int q0  = (wid % (NQ / 16)) * 16;
  int lane = threadIdx.x & 31;
  int l = lane & 15, g = lane >> 4;

  const _Float16* Qbh = Qt + (size_t)bh * D_ * NQ;
  const _Float16* Kbh = Kb + (size_t)bh * NKV * D_;
  const _Float16* Vbh = Vb + (size_t)bh * NKV * D_;

  // Q^T as B fragments (64 = two 32-deep k-steps), resident for whole loop
  v16h bq0 = load_b_frag(Qbh, NQ, 0,  q0, lane);
  v16h bq1 = load_b_frag(Qbh, NQ, 32, q0, lane);

  float m = -1e30f, lsum = 0.f;       // per lane: stats for row q = l (mirrored halves)
  v8f accO[4] = {};                    // O^... : (M=q, N=d) 4 d-tiles of 16

  for (int kt = 0; kt < NKV; kt += 32) {
    // St0: kv rows kt..kt+15 ; St1: kv rows kt+16..kt+31
    v8f st0 = {}, st1 = {};
    st0 = wmma_f16(load_a_frag(Kbh, D_, kt,      0,  lane), bq0, st0);
    st0 = wmma_f16(load_a_frag(Kbh, D_, kt,      32, lane), bq1, st0);
    st1 = wmma_f16(load_a_frag(Kbh, D_, kt + 16, 0,  lane), bq0, st1);
    st1 = wmma_f16(load_a_frag(Kbh, D_, kt + 16, 32, lane), bq1, st1);
    // (first softmax scale already folded into Qt)

    float mx = -1e30f;
#pragma unroll
    for (int v = 0; v < 8; ++v) mx = fmaxf(mx, fmaxf(st0[v], st1[v]));
    mx = fmaxf(mx, __shfl_xor(mx, 16));          // combine half-waves: all 32 kv
    float mnew = fmaxf(m, mx);

    float rs = 0.f;
    v16h pa;                                     // P as A fragment (16q x 32kv)
#pragma unroll
    for (int v = 0; v < 8; ++v) {
      float p0 = __expf(st0[v] - mnew);
      float p1 = __expf(st1[v] - mnew);
      rs += p0 + p1;
      pa[v]     = (_Float16)p0;                  // halves 0..7  -> kv = 8g+v
      pa[v + 8] = (_Float16)p1;                  // halves 8..15 -> kv = 16+8g+v
    }
    rs += __shfl_xor(rs, 16);
    float corr = __expf(m - mnew);
    lsum = lsum * corr + rs;
    m = mnew;

    // rescale O accumulators: reg v holds row q = v + 8g -> fetch corr[q]
#pragma unroll
    for (int v = 0; v < 8; ++v) {
      float cw = __shfl(corr, v + 8 * g);
#pragma unroll
      for (int t = 0; t < 4; ++t) accO[t][v] *= cw;
    }

    // O += P * V  (V row-major; B fragment = strided per-lane loads)
#pragma unroll
    for (int t = 0; t < 4; ++t) {
      v16h bv = load_b_frag(Vbh, D_, kt, t * 16, lane);
      accO[t] = wmma_f16(pa, bv, accO[t]);
    }
  }

  // normalize (and apply the reference's quirk second *scale), store f16 X
  float fin = SCALE / lsum;
  int b = bh / H_, hh = bh % H_;
#pragma unroll
  for (int v = 0; v < 8; ++v) {
    float fv = __shfl(fin, v + 8 * g);
    int q = q0 + v + 8 * g;
    _Float16* row = X + ((size_t)(b * NQ + q)) * C_ + hh * D_;
#pragma unroll
    for (int t = 0; t < 4; ++t) row[t * 16 + l] = (_Float16)(accO[t][v] * fv);
  }
}

// ---------------------------------------------------------------------------
// Output projection: X[4096x768] * Wproj16[768x768] + bproj -> fp32 out
// ---------------------------------------------------------------------------
__global__ void gemm_out_kernel(const _Float16* __restrict__ A, const _Float16* __restrict__ W,
                                const float* __restrict__ bias, float* __restrict__ out) {
  const int N = C_, K = C_;
  int wid  = blockIdx.x * (blockDim.x >> 5) + (threadIdx.x >> 5);
  int tnn  = N / 32;
  int row0 = (wid / tnn) * 32, col0 = (wid % tnn) * 32;
  int lane = threadIdx.x & 31;
  v8f acc[2][2] = {};
  gemm_core(A, W, K, N, row0, col0, lane, acc);

  int l = lane & 15, g = lane >> 4;
#pragma unroll
  for (int rt = 0; rt < 2; ++rt)
#pragma unroll
    for (int ct = 0; ct < 2; ++ct) {
      int c = col0 + ct * 16 + l;
      float bc = bias[c];
#pragma unroll
      for (int v = 0; v < 8; ++v) {
        int r = row0 + rt * 16 + 8 * g + v;
        out[(size_t)r * C_ + c] = acc[rt][ct][v] + bc;
      }
    }
}

// ---------------------------------------------------------------------------
extern "C" void kernel_launch(void* const* d_in, const int* in_sizes, int n_in,
                              void* d_out, int out_size, void* d_ws, size_t ws_size,
                              hipStream_t stream) {
  const float* q_token  = (const float*)d_in[0];
  const float* kv_token = (const float*)d_in[1];
  const float* Wq       = (const float*)d_in[2];
  const float* Wkv      = (const float*)d_in[3];
  const float* Wproj    = (const float*)d_in[4];
  const float* bproj    = (const float*)d_in[5];
  float* out = (float*)d_out;

  // workspace carve-up (f16), ~42.5 MB total
  _Float16* p = (_Float16*)d_ws;
  _Float16* q16     = p; p += (size_t)B_ * NQ * C_;
  _Float16* kv16    = p; p += (size_t)B_ * NKV * C_;
  _Float16* Wq16    = p; p += (size_t)C_ * C_;
  _Float16* Wkv16   = p; p += (size_t)C_ * 2 * C_;
  _Float16* Wproj16 = p; p += (size_t)C_ * C_;
  _Float16* Qt      = p; p += (size_t)B_ * H_ * D_ * NQ;
  _Float16* Kb      = p; p += (size_t)B_ * H_ * NKV * D_;
  _Float16* Vb      = p; p += (size_t)B_ * H_ * NKV * D_;
  _Float16* Xb      = p; p += (size_t)B_ * NQ * C_;

  auto cvt = [&](const float* s, _Float16* d, int n) {
    int t = n / 4;
    cvt_kernel<<<(t + 255) / 256, 256, 0, stream>>>(s, d, n);
  };
  cvt(q_token,  q16,  B_ * NQ * C_);
  cvt(kv_token, kv16, B_ * NKV * C_);
  cvt(Wq,    Wq16,    C_ * C_);
  cvt(Wkv,   Wkv16,   C_ * 2 * C_);
  cvt(Wproj, Wproj16, C_ * C_);

  // Q proj: (4096/32)*(768/32)=3072 waves, 4 waves/block
  gemm_q_kernel<<<768, 128, 0, stream>>>(q16, Wq16, Qt);
  // KV proj: (4096/32)*(1536/32)=6144 waves
  gemm_kv_kernel<<<1536, 128, 0, stream>>>(kv16, Wkv16, Kb, Vb);
  // attention: B*H*(NQ/16)=3072 waves
  attn_kernel<<<768, 128, 0, stream>>>(Qt, Kb, Vb, Xb);
  // out proj
  gemm_out_kernel<<<768, 128, 0, stream>>>(Xb, Wproj16, bproj, out);
}
